// CrossViewSwapAttention_63642825392268
// MI455X (gfx1250) — matrix operand order repaired and checked by reading, not compiled, not run on hardware
//
#include <hip/hip_runtime.h>
#include <hip/hip_bf16.h>

// ---------------------------------------------------------------------------
// CrossViewSwapAttention fused kernel for gfx1250 (MI455X, wave32, WMMA)
// One workgroup (256 threads = 8 waves) per spatial position l (625 blocks).
// All GEMMs via v_wmma_f32_16x16x32_f16.
// K is stored head-transposed (KT[(m*32+e)][token]) so every B-fragment in
// the attention phase is read row-major in k via loadB_direct (16-bit strided loads).
// ---------------------------------------------------------------------------

typedef _Float16 f16;
typedef __attribute__((ext_vector_type(16))) _Float16 v16h;
typedef __attribute__((ext_vector_type(8)))  float    v8f;

#define NVIEW 6
#define LPOS  625      // X*Y = 25*25
#define QN    384      // N*W1*W2
#define KN    96       // N*w1*w2
#define DMODEL 128
#define HEADS 4
#define DH    32
#define LOG_INV_SQRT_DH (-1.7328679513998633f)  // ln(32^-0.5)

// LDS layout (in f16 units)
#define OFF_W    0                    // phase1: Wq/Wk/Wv [3][128][128] ; phase2/3: a-buffer [384][128]
#define OFF_QH   49152                // qh [384][128]
#define OFF_KT   (49152 + 49152)      // KT [128][96]  (phase3: wprojT [128][128] starts here)
#define OFF_VH   (OFF_KT + 12288)     // vh [96][128]  (phase3: abar [64][128] at OFF_KT+16384)
#define OFF_WS   (OFF_VH + 12288)     // per-wave scratch: 8 * 2048 f16 (4KB each)
#define SMEM_F16 (OFF_WS + 8 * 2048)  // 139264 f16 = 278528 bytes

union Frag {
    v16h v;
    unsigned int u[8];
    f16 h[16];
};

__device__ __forceinline__ int kbaseA(int g, int hi) {
    return ((g >> 2) << 4) + (hi << 3) + ((g & 3) << 1);
}

// A-fragment: 16x32 f16 tile, src row-major [16][S], column offset c0 (even).
__device__ __forceinline__ v16h loadA(const f16* __restrict__ base, int S, int c0, int lane) {
    int row = lane & 15, hi = lane >> 4;
    Frag fr;
#pragma unroll
    for (int g = 0; g < 8; ++g)
        fr.u[g] = *(const unsigned int*)(base + row * S + c0 + kbaseA(g, hi));
    return fr.v;
}

// B-fragment (32x16), B[kk][n] = src[(kk0+kk)*S + n0 + n]  (row-major in K)
__device__ __forceinline__ v16h loadB_direct(const f16* __restrict__ src, int S, int kk0, int n0, int lane) {
    const f16* p = src + kk0 * S + n0 + (lane & 15);
    const int hi = lane >> 4;
    Frag fr;
#pragma unroll
    for (int g = 0; g < 8; ++g) {
        fr.h[2 * g]     = p[(kbaseA(g, hi)) * S];
        fr.h[2 * g + 1] = p[(kbaseA(g, hi) + 1) * S];
    }
    return fr.v;
}

__device__ __forceinline__ v8f wmma_f16(v16h a, v16h b, v8f c) {
    return __builtin_amdgcn_wmma_f32_16x16x32_f16(false, a, false, b, (short)0, c, false, false);
}

// Store C (16x16 f32 acc) into row-major f16 [rows][S] at (r0, c0)
__device__ __forceinline__ void storeC_f16(f16* __restrict__ dst, int S, int r0, int c0, v8f acc, int lane) {
    int c = lane & 15, hi = lane >> 4;
#pragma unroll
    for (int r = 0; r < 8; ++r)
        dst[(r0 + r + 8 * hi) * S + c0 + c] = (f16)acc[r];
}

// Store C transposed: dst[(c0 + col)][r0 + row], row stride S  (for KT)
__device__ __forceinline__ void storeC_f16T(f16* __restrict__ dst, int S, int c0, int r0, v8f acc, int lane) {
    int c = lane & 15, hi = lane >> 4;
#pragma unroll
    for (int r = 0; r < 8; ++r)
        dst[(c0 + c) * S + r0 + r + 8 * hi] = (f16)acc[r];
}

__global__ void __launch_bounds__(256, 1)
cvswap_attn_kernel(const float* __restrict__ q, const float* __restrict__ k,
                   const float* __restrict__ v, const float* __restrict__ skip,
                   const float* __restrict__ rope, const float* __restrict__ gate,
                   const float* __restrict__ lnqg, const float* __restrict__ lnqb,
                   const float* __restrict__ lnkg, const float* __restrict__ lnkb,
                   const float* __restrict__ lnvg, const float* __restrict__ lnvb,
                   const float* __restrict__ wq, const float* __restrict__ bq,
                   const float* __restrict__ wk, const float* __restrict__ bk,
                   const float* __restrict__ wv, const float* __restrict__ bv,
                   const float* __restrict__ wproj, const float* __restrict__ bproj,
                   const float* __restrict__ als, float* __restrict__ out) {
    extern __shared__ __align__(16) char smem_raw[];
    f16* S0 = reinterpret_cast<f16*>(smem_raw);

    const int l    = blockIdx.x;       // spatial position 0..624
    const int tid  = threadIdx.x;
    const int lane = tid & 31;
    const int wid  = tid >> 5;

    f16* WREG = S0 + OFF_W;            // [3][128][128]  (phase1)
    f16* ABUF = S0 + OFF_W;            // [384][128]     (phase2/3, reuses WREG)
    f16* QH   = S0 + OFF_QH;           // [384][128]
    f16* KT   = S0 + OFF_KT;           // [128][96]  rows = (head,e), cols = k-token
    f16* VH   = S0 + OFF_VH;           // [96][128]
    f16* WSC  = S0 + OFF_WS + wid * 2048;  // per-wave scratch, 4KB

    // ---------------- Phase 0: stage Wq/Wk/Wv as f16, transposed [d][o] ---------------
    for (int i = tid; i < 3 * 16384; i += 256) {
        int kind = i >> 14;
        int r = i & 16383;
        int o = r >> 7, d = r & 127;
        const float* W = (kind == 0) ? wq : (kind == 1) ? wk : wv;
        WREG[kind * 16384 + d * 128 + o] = (f16)W[o * 128 + d];
    }
    __syncthreads();

    // ---------------- Phase 1: LayerNorm + projection + RoPE -> qh/KT/vh --------------
    // 36 tiles of 16 tokens: 24 q-tiles, 6 k-tiles, 6 v-tiles
    for (int tt = wid; tt < 36; tt += 8) {
        const int kind  = (tt < 24) ? 0 : (tt < 30 ? 1 : 2);
        const int tile  = (kind == 0) ? tt : (kind == 1 ? tt - 24 : tt - 30);
        const int tbase = tile * 16;
        const float* gx   = (kind == 0) ? q : (kind == 1) ? k : v;
        const float* lng  = (kind == 0) ? lnqg : (kind == 1) ? lnkg : lnvg;
        const float* lnb  = (kind == 0) ? lnqb : (kind == 1) ? lnkb : lnvb;
        const float* bias = (kind == 0) ? bq : (kind == 1) ? bk : bv;
        const f16*   W    = WREG + kind * 16384;

        const float4 gmv = *(const float4*)(lng + 4 * lane);
        const float4 btv = *(const float4*)(lnb + 4 * lane);

        // LayerNorm 16 tokens into wave tile WSC [16][128] f16
        for (int tok = 0; tok < 16; ++tok) {
            int t = tbase + tok;
            long row;
            if (kind == 0) { int n = t >> 6; row = ((long)n * LPOS + l) * 64 + (t & 63); }
            else           { int n = t >> 4; row = ((long)n * LPOS + l) * 16 + (t & 15); }
            const float* src = gx + row * DMODEL;
            float4 xv = *(const float4*)(src + 4 * lane);
            float s = xv.x + xv.y + xv.z + xv.w;
#pragma unroll
            for (int m = 16; m >= 1; m >>= 1) s += __shfl_xor(s, m, 32);
            float mu = s * (1.0f / 128.0f);
            float d0 = xv.x - mu, d1 = xv.y - mu, d2 = xv.z - mu, d3 = xv.w - mu;
            float ss = d0 * d0 + d1 * d1 + d2 * d2 + d3 * d3;
#pragma unroll
            for (int m = 16; m >= 1; m >>= 1) ss += __shfl_xor(ss, m, 32);
            float rstd = rsqrtf(ss * (1.0f / 128.0f) + 1e-5f);
            union { f16 h[4]; unsigned long long ull; } pk;
            pk.h[0] = (f16)(d0 * rstd * gmv.x + btv.x);
            pk.h[1] = (f16)(d1 * rstd * gmv.y + btv.y);
            pk.h[2] = (f16)(d2 * rstd * gmv.z + btv.z);
            pk.h[3] = (f16)(d3 * rstd * gmv.w + btv.w);
            *(unsigned long long*)(WSC + tok * 128 + 4 * lane) = pk.ull;
        }
        __builtin_amdgcn_wave_barrier();
        asm volatile("s_wait_dscnt 0x0" ::: "memory");

        // A-fragments (shared across all output tiles): 4 K-chunks of 32
        v16h afr[4];
#pragma unroll
        for (int kc = 0; kc < 4; ++kc) afr[kc] = loadA(WSC, 128, kc * 32, lane);

        const int c = lane & 15, hi = lane >> 4;
        // 8 output n-tiles processed as 4 pairs (RoPE couples e <-> e+16)
        for (int p = 0; p < 4; ++p) {
            float b0 = bias[p * 32 + c], b1 = bias[p * 32 + 16 + c];
            v8f acc0, acc1;
#pragma unroll
            for (int r = 0; r < 8; ++r) { acc0[r] = b0; acc1[r] = b1; }
#pragma unroll
            for (int kc = 0; kc < 4; ++kc) {
                v16h bA = loadB_direct(W, 128, kc * 32, p * 32, lane);
                acc0 = wmma_f16(afr[kc], bA, acc0);
                v16h bB = loadB_direct(W, 128, kc * 32, p * 32 + 16, lane);
                acc1 = wmma_f16(afr[kc], bB, acc1);
            }
            if (kind < 2) {
                // RoPE (+ fold per-head temp*gate into q)
                float scl = 1.0f;
                if (kind == 0) {
                    float tmp = als[p] + LOG_INV_SQRT_DH;
                    float g   = fminf(fmaxf(gate[p], 0.0f), 1.0f);
                    scl = tmp * g;
                }
#pragma unroll
                for (int r = 0; r < 8; ++r) {
                    int t = tbase + r + 8 * hi;
                    float f0 = rope[t * 32 + c];
                    float f1 = rope[t * 32 + c + 16];
                    float s0, c0s, s1, c1s;
                    __sincosf(f0, &s0, &c0s);
                    __sincosf(f1, &s1, &c1s);
                    float x0 = acc0[r], x1 = acc1[r];
                    acc0[r] = (x0 * c0s - x1 * s0) * scl;
                    acc1[r] = (x1 * c1s + x0 * s1) * scl;
                }
            }
            if (kind == 0) {
                storeC_f16(QH, 128, tbase, p * 32, acc0, lane);
                storeC_f16(QH, 128, tbase, p * 32 + 16, acc1, lane);
            } else if (kind == 1) {
                // transposed store: KT[(m*32+e)][token], row stride 96
                storeC_f16T(KT, 96, p * 32, tbase, acc0, lane);
                storeC_f16T(KT, 96, p * 32 + 16, tbase, acc1, lane);
            } else {
                storeC_f16(VH, 128, tbase, p * 32, acc0, lane);
                storeC_f16(VH, 128, tbase, p * 32 + 16, acc1, lane);
            }
        }
    }
    __syncthreads();

    // ---------------- Phase 2: attention (per head m, per 16-row q-tile) --------------
    // 96 units = 4 heads * 24 q-tiles ; 12 per wave. Output -> ABUF (reuses W region).
    for (int u = wid; u < 96; u += 8) {
        const int m  = u / 24;
        const int qt = u - m * 24;
        const v16h qa = loadA(QH + qt * 16 * 128, 128, m * 32, lane);

        v8f s[6];
#pragma unroll
        for (int kt = 0; kt < 6; ++kt) {
            v8f z;
#pragma unroll
            for (int r = 0; r < 8; ++r) z[r] = 0.0f;
            // B = K^T directly from KT: B[kk][n] = KT[(m*32+kk)*96 + kt*16 + n]
            v16h bK = loadB_direct(KT, 96, m * 32, kt * 16, lane);
            s[kt] = wmma_f16(qa, bK, z);
        }
        // softmax over 96 columns; row lives across a 16-lane group
#pragma unroll
        for (int r = 0; r < 8; ++r) {
            float mx = s[0][r];
#pragma unroll
            for (int kt = 1; kt < 6; ++kt) mx = fmaxf(mx, s[kt][r]);
#pragma unroll
            for (int msk = 8; msk >= 1; msk >>= 1) mx = fmaxf(mx, __shfl_xor(mx, msk, 32));
            float sum = 0.0f;
#pragma unroll
            for (int kt = 0; kt < 6; ++kt) {
                float e = __expf(s[kt][r] - mx);
                s[kt][r] = e;
                sum += e;
            }
#pragma unroll
            for (int msk = 8; msk >= 1; msk >>= 1) sum += __shfl_xor(sum, msk, 32);
            float inv = 1.0f / sum;
#pragma unroll
            for (int kt = 0; kt < 6; ++kt) s[kt][r] *= inv;
        }
        // stage P (16x96 f16) in wave scratch
#pragma unroll
        for (int kt = 0; kt < 6; ++kt) storeC_f16(WSC, 96, 0, kt * 16, s[kt], lane);
        __builtin_amdgcn_wave_barrier();
        asm volatile("s_wait_dscnt 0x0" ::: "memory");

        // O = P @ V  (16x32), K=96 in 3 chunks
#pragma unroll
        for (int nt = 0; nt < 2; ++nt) {
            v8f acc;
#pragma unroll
            for (int r = 0; r < 8; ++r) acc[r] = 0.0f;
#pragma unroll
            for (int kc = 0; kc < 3; ++kc) {
                v16h pa = loadA(WSC, 96, kc * 32, lane);
                v16h vb = loadB_direct(VH, 128, kc * 32, m * 32 + nt * 16, lane);
                acc = wmma_f16(pa, vb, acc);
            }
            storeC_f16(ABUF, 128, qt * 16, m * 32 + nt * 16, acc, lane);
        }
    }
    __syncthreads();

    // ---------------- Phase 3: mean over views + out-projection + skip ----------------
    f16* WPT = S0 + OFF_KT;            // wprojT [128][128] (reuses KT + part of vh)
    f16* AB  = WPT + 16384;            // abar [64][128]
    for (int i = tid; i < 16384; i += 256) {
        int o = i >> 7, c2 = i & 127;
        WPT[c2 * 128 + o] = (f16)wproj[o * 128 + c2];
    }
    for (int i = tid; i < 8192; i += 256) {
        int uv = i >> 7, c2 = i & 127;
        float sum = 0.0f;
#pragma unroll
        for (int n = 0; n < NVIEW; ++n) sum += (float)ABUF[(n * 64 + uv) * 128 + c2];
        AB[uv * 128 + c2] = (f16)(sum * (1.0f / 6.0f));
    }
    __syncthreads();

    // Z (64x128) = abar @ wproj^T : 4 row-tiles * 8 n-tiles = 32 units, 4 per wave
    for (int u2 = wid; u2 < 32; u2 += 8) {
        const int rt = u2 & 3, nt = u2 >> 2;
        v8f acc;
#pragma unroll
        for (int r = 0; r < 8; ++r) acc[r] = 0.0f;
#pragma unroll
        for (int kc = 0; kc < 4; ++kc) {
            v16h aa = loadA(AB + rt * 16 * 128, 128, kc * 32, lane);
            v16h bb = loadB_direct(WPT, 128, kc * 32, nt * 16, lane);
            acc = wmma_f16(aa, bb, acc);
        }
        const int c = lane & 15, hi = lane >> 4;
        const int o = nt * 16 + c;
        const float bpv = bproj[o];
#pragma unroll
        for (int r = 0; r < 8; ++r) {
            int uv = rt * 16 + r + 8 * hi;
            size_t idx = ((size_t)l * 64 + uv) * DMODEL + o;
            out[idx] = acc[r] + bpv + skip[idx];
        }
    }
}

extern "C" void kernel_launch(void* const* d_in, const int* in_sizes, int n_in,
                              void* d_out, int out_size, void* d_ws, size_t ws_size,
                              hipStream_t stream) {
    (void)in_sizes; (void)n_in; (void)d_ws; (void)ws_size; (void)out_size;
    const float* q     = (const float*)d_in[0];
    const float* k     = (const float*)d_in[1];
    const float* v     = (const float*)d_in[2];
    const float* skip  = (const float*)d_in[3];
    const float* rope  = (const float*)d_in[4];
    const float* gate  = (const float*)d_in[5];
    const float* lnqg  = (const float*)d_in[6];
    const float* lnqb  = (const float*)d_in[7];
    const float* lnkg  = (const float*)d_in[8];
    const float* lnkb  = (const float*)d_in[9];
    const float* lnvg  = (const float*)d_in[10];
    const float* lnvb  = (const float*)d_in[11];
    const float* wq    = (const float*)d_in[12];
    const float* bq    = (const float*)d_in[13];
    const float* wk    = (const float*)d_in[14];
    const float* bk    = (const float*)d_in[15];
    const float* wv    = (const float*)d_in[16];
    const float* bv    = (const float*)d_in[17];
    const float* wproj = (const float*)d_in[18];
    const float* bproj = (const float*)d_in[19];
    const float* als   = (const float*)d_in[20];
    float* out = (float*)d_out;

    const size_t shmem = (size_t)SMEM_F16 * sizeof(_Float16);  // 278,528 B (< 320 KB/WGP)
    (void)hipFuncSetAttribute((const void*)cvswap_attn_kernel,
                              hipFuncAttributeMaxDynamicSharedMemorySize, (int)shmem);
    cvswap_attn_kernel<<<LPOS, 256, shmem, stream>>>(
        q, k, v, skip, rope, gate, lnqg, lnqb, lnkg, lnkb, lnvg, lnvb,
        wq, bq, wk, bk, wv, bv, wproj, bproj, als, out);
}
